// ATDTransformerLayer_13846974562650
// MI455X (gfx1250) — matrix operand order repaired and checked by reading, not compiled, not run on hardware
//
#include <hip/hip_runtime.h>
#include <math.h>

#if __has_include(<hip/amd_detail/amd_gfx1250_TDM.h>)
#define ATH_TDM6 1          // therock-10.0 headers -> 6-arg tensor builtin
#else
#define ATH_TDM6 0          // ROCm 7.2 -> 5-arg tensor builtin
#endif
#define ATH_USE_TDM 1

typedef __attribute__((ext_vector_type(16))) __bf16 v16bf;
typedef __attribute__((ext_vector_type(8)))  __bf16 v8bf;
typedef __attribute__((ext_vector_type(8)))  float  v8f;
typedef __attribute__((ext_vector_type(4)))  uint32_t u32x4;
typedef __attribute__((ext_vector_type(8)))  int      i32x8;
typedef __attribute__((ext_vector_type(4)))  int      i32x4;

#define DEV __device__ __forceinline__

constexpr int Bn   = 4;
constexpr int Hh   = 128, Wwid = 128;
constexpr int Nn   = Hh * Wwid;        // 16384 tokens per batch
constexpr int DIM  = 210, DIMP = 224;  // channel dim, K-padded to 32-multiple
constexpr int C3   = 3 * DIM;          // 630
constexpr int HEADS = 6, HD = 35;
constexpr int DR   = 20, Mtd = 128;
constexpr int MLP  = 420, DTD = 96, HF2 = 516, HF2P = 544;
// N-padded (multiple of 64) weight leading dims so B-tile loads need no bounds checks
constexpr int C3N  = 640, DIMN = 256, DTDN = 128, MLPN = 448;

DEV float geluf(float x) { return 0.5f * x * (1.0f + erff(x * 0.7071067811865476f)); }

// ---------------------------------------------------------------------------
// Tensor Data Mover: async 2-D tile load (rows x rowlen bf16) into LDS.
// D# built per CDNA5 ISA section 8 (group0/group1 bitfields), count=1,
// data_size=2B, tile_dim0=rowlen, tile_dim1=rows, tensor_dim0_stride=stride.
// ---------------------------------------------------------------------------
DEV void tdm_load_2d_bf16(const __bf16* gsrc, uint32_t lds_off, int rows, int rowlen,
                          long stride_elems)
{
    uint64_t ga = (uint64_t)(uintptr_t)gsrc;
    u32x4 g0;
    g0[0] = 1u;                                           // count=1, user descriptor
    g0[1] = lds_off;                                      // lds_addr (bytes)
    g0[2] = (uint32_t)(ga & 0xffffffffu);                 // global_addr[31:0]
    g0[3] = (uint32_t)((ga >> 32) & 0x01ffffffu) | (2u << 30);  // addr[56:32] | type=2
    const uint32_t td0 = 0x40000000u, td1 = 0x40000000u;  // huge tensor dims: no OOB clip
    i32x8 g1;
    g1[0] = (int)(1u << 16);                              // data_size = 2 bytes
    g1[1] = (int)((td0 & 0xffffu) << 16);                 // tensor_dim0[15:0]
    g1[2] = (int)(((td0 >> 16) & 0xffffu) | ((td1 & 0xffffu) << 16));
    g1[3] = (int)(((td1 >> 16) & 0xffffu) | (((uint32_t)rowlen & 0xffffu) << 16)); // tile_dim0
    g1[4] = (int)((uint32_t)rows & 0xffffu);              // tile_dim1 (tile_dim2=0)
    uint64_t s0 = (uint64_t)stride_elems;                 // tensor_dim0_stride (elems)
    g1[5] = (int)(uint32_t)(s0 & 0xffffffffu);
    g1[6] = (int)(uint32_t)((s0 >> 32) & 0xffffu);
    g1[7] = 0;
    i32x4 gz = {0, 0, 0, 0};
#if ATH_TDM6
    i32x8 gz8 = {0, 0, 0, 0, 0, 0, 0, 0};
    __builtin_amdgcn_tensor_load_to_lds(g0, g1, gz, gz, gz8, 0);
#else
    __builtin_amdgcn_tensor_load_to_lds(g0, g1, gz, gz, 0);
#endif
}

// ---------------------------------------------------------------------------
// f32 -> bf16 convert with 2-D zero padding. dst [rowsPad x ldd].
// ---------------------------------------------------------------------------
__global__ void cvt_pad_kernel(const float* __restrict__ src, __bf16* __restrict__ dst,
                               int rows, int cols, int ldd, int rowsPad)
{
    long i = (long)blockIdx.x * 256 + threadIdx.x;
    long tot = (long)rowsPad * ldd;
    if (i >= tot) return;
    int r = (int)(i / ldd), c = (int)(i % ldd);
    float v = (r < rows && c < cols) ? src[(long)r * cols + c] : 0.0f;
    dst[i] = (__bf16)v;
}

// ---------------------------------------------------------------------------
// LayerNorm over 210 channels, one wave per token, writes bf16 padded to 224.
// ---------------------------------------------------------------------------
__global__ __launch_bounds__(256) void ln_kernel(const float* __restrict__ x,
                                                 const float* __restrict__ g,
                                                 const float* __restrict__ bvec,
                                                 __bf16* __restrict__ out, int ntok)
{
    int wid = threadIdx.x >> 5, lane = threadIdx.x & 31;
    int tok = blockIdx.x * 8 + wid;
    const float* xp = x + (long)tok * DIM;
    float s = 0.f, s2 = 0.f, vals[7];
    #pragma unroll
    for (int i = 0; i < 7; ++i) {
        int c = lane + 32 * i;
        float v = (c < DIM) ? xp[c] : 0.f;
        vals[i] = v; s += v; s2 += v * v;
    }
    #pragma unroll
    for (int m = 1; m < 32; m <<= 1) { s += __shfl_xor(s, m, 32); s2 += __shfl_xor(s2, m, 32); }
    float mu = s / (float)DIM;
    float var = s2 / (float)DIM - mu * mu;
    float rs = rsqrtf(var + 1e-5f);
    __bf16* op = out + (long)tok * DIMP;
    #pragma unroll
    for (int i = 0; i < 7; ++i) {
        int c = lane + 32 * i;
        if (c < DIMP) {
            float v = (c < DIM) ? (vals[i] - mu) * rs * g[c] + bvec[c] : 0.f;
            op[c] = (__bf16)v;
        }
    }
    (void)ntok;
}

// ---------------------------------------------------------------------------
// Fragment helpers (wave32 WMMA bf16 layouts).
// ---------------------------------------------------------------------------
DEV v16bf frag_a(const __bf16* base /* row start */, int lh)
{
    v8bf a0 = *(const v8bf*)(base + lh * 8);
    v8bf a1 = *(const v8bf*)(base + 16 + lh * 8);
    return __builtin_shufflevector(a0, a1, 0, 1, 2, 3, 4, 5, 6, 7,
                                   8, 9, 10, 11, 12, 13, 14, 15);
}
DEV v16bf frag_b_sw(const __bf16* swz, int tile, int lane)
{
    return *(const v16bf*)(swz + ((long)tile * 32 + lane) * 16);
}
DEV int bswz_addr(int tile, int k, int colInTile)
{
    return ((tile * 32 + (((k >> 3) & 1) << 4) + colInTile) << 4) + ((k >> 4) << 3) + (k & 7);
}

// ---------------------------------------------------------------------------
// Generic bf16 WMMA GEMM: C[M,N] = A[M,K] @ W[K,N] (+bias)(+addsrc)(GELU)(TRANS)
// K % 32 == 0, M % 64 == 0, ldw % 64 == 0 (buffers pre-padded).
// Double-buffered: TDM DMA + B global load of step s+1 overlap WMMAs of step s.
// ---------------------------------------------------------------------------
template<bool OUTB, bool ADDSRC, bool GELU_, bool TRANS>
__global__ __launch_bounds__(256) void gemm_bf16_wmma(
    const __bf16* __restrict__ A, long sAb, int lda,
    const __bf16* __restrict__ Wm, long sWb, int ldw,
    const float* __restrict__ bias,
    const float* __restrict__ asrc, long sSb, int lds_,
    float* __restrict__ outF, __bf16* __restrict__ outB, long sCb, int ldc,
    int M, int N, int K)
{
    __shared__ __bf16 As[2][64 * 32];        // natural row-major, ping-pong
    __shared__ __bf16 Bsw[2][4 * 32 * 16];   // swizzled fragment order, ping-pong
    const int tid = threadIdx.x, lane = tid & 31, wid = tid >> 5;
    const int l4 = lane & 15, lh = lane >> 4;
    const int bz = blockIdx.z;
    const __bf16* Ab = A + (long)bz * sAb;
    const __bf16* Wb = Wm + (long)bz * sWb;
    const int row0 = blockIdx.y * 64, col0 = blockIdx.x * 64;
    const int wm = wid & 1, wn = wid >> 1;
    const int br = tid >> 3;               // B fill: k row within 32-block
    const int bc8 = (tid & 7) * 8;         // B fill: col chunk within 64
    v8f acc0 = {}; v8f acc1 = {};
    const int nsteps = K >> 5;
#if ATH_USE_TDM
    const uint32_t as_off0 = (uint32_t)(uintptr_t)(&As[0][0]);
    const uint32_t as_off1 = (uint32_t)(uintptr_t)(&As[1][0]);
    // ---- prologue: stage step 0 ----
    if (wid == 0) {
        tdm_load_2d_bf16(&Ab[(long)row0 * lda], as_off0, 64, 32, lda);
    }
#else
    {
        int r = tid >> 2, c8 = (tid & 3) * 8;
        *(v8bf*)&As[0][r * 32 + c8] = *(const v8bf*)&Ab[(long)(row0 + r) * lda + c8];
    }
#endif
    {
        v8bf w8 = *(const v8bf*)&Wb[(long)br * ldw + col0 + bc8];
        #pragma unroll
        for (int j = 0; j < 8; ++j)
            Bsw[0][bswz_addr((bc8 + j) >> 4, br, (bc8 + j) & 15)] = w8[j];
    }
#if ATH_USE_TDM
    if (wid == 0) __builtin_amdgcn_s_wait_tensorcnt(0);
#endif
    __syncthreads();

    for (int s = 0; s < nsteps; ++s) {
        const int cur = s & 1, nxt = cur ^ 1;
        // ---- stage step s+1 into the alternate buffers ----
        if (s + 1 < nsteps) {
            const int k1 = (s + 1) << 5;
#if ATH_USE_TDM
            if (wid == 0)
                tdm_load_2d_bf16(&Ab[(long)row0 * lda + k1], nxt ? as_off1 : as_off0,
                                 64, 32, lda);
#else
            {
                int r = tid >> 2, c8 = (tid & 3) * 8;
                *(v8bf*)&As[nxt][r * 32 + c8] =
                    *(const v8bf*)&Ab[(long)(row0 + r) * lda + k1 + c8];
            }
#endif
            v8bf w8 = *(const v8bf*)&Wb[(long)(k1 + br) * ldw + col0 + bc8];
            #pragma unroll
            for (int j = 0; j < 8; ++j)
                Bsw[nxt][bswz_addr((bc8 + j) >> 4, br, (bc8 + j) & 15)] = w8[j];
            if (s + 2 < nsteps)
                __builtin_prefetch(&Wb[(long)(((s + 2) << 5) + (tid & 31)) * ldw + col0]);
        }
        // ---- compute step s (overlaps the in-flight DMA / global load) ----
        v16bf bfr = frag_b_sw(Bsw[cur], wn, lane);
        #pragma unroll
        for (int t = 0; t < 2; ++t) {
            v16bf af = frag_a(&As[cur][(wm * 32 + t * 16 + l4) * 32], lh);
            if (t == 0)
                acc0 = __builtin_amdgcn_wmma_f32_16x16x32_bf16(false, af, false, bfr, (short)0, acc0, false, false);
            else
                acc1 = __builtin_amdgcn_wmma_f32_16x16x32_bf16(false, af, false, bfr, (short)0, acc1, false, false);
        }
#if ATH_USE_TDM
        if (wid == 0 && s + 1 < nsteps) __builtin_amdgcn_s_wait_tensorcnt(0);
#endif
        __syncthreads();
    }
    const int col = col0 + wn * 16 + l4;
    if (col >= N) return;
    const float bv = bias ? bias[col] : 0.0f;
    #pragma unroll
    for (int t = 0; t < 2; ++t) {
        #pragma unroll
        for (int r = 0; r < 8; ++r) {
            int row = row0 + wm * 32 + t * 16 + lh * 8 + r;   // M % 64 == 0
            float v = (t ? acc1[r] : acc0[r]) + bv;
            if (GELU_)  v = geluf(v);
            if (ADDSRC) v += asrc[(long)bz * sSb + (long)row * lds_ + col];
            if (TRANS)       outF[(long)bz * sCb + (long)col * ldc + row] = v;
            else if (OUTB)   outB[(long)bz * sCb + (long)row * ldc + col] = (__bf16)v;
            else             outF[(long)bz * sCb + (long)row * ldc + col] = v;
        }
    }
}

// ---------------------------------------------------------------------------
// l2-normalized K projection of the token dictionary: one wave per (b,m).
// ---------------------------------------------------------------------------
__global__ __launch_bounds__(256) void knorm_kernel(const float* __restrict__ td,
                                                    const float* __restrict__ wk,
                                                    const float* __restrict__ kb,
                                                    float* __restrict__ out)
{
    int wid = threadIdx.x >> 5, lane = threadIdx.x & 31;
    int row = blockIdx.x * 8 + wid;          // 512 rows exactly
    __shared__ float ts[8][DIM];
    const float* tp = td + (long)row * DIM;
    for (int i = lane; i < DIM; i += 32) ts[wid][i] = tp[i];
    __syncthreads();
    float kv = 0.f;
    if (lane < DR) {
        for (int c = 0; c < DIM; ++c) kv += ts[wid][c] * wk[c * DR + lane];
        kv += kb[lane];
    }
    float s2 = (lane < DR) ? kv * kv : 0.f;
    #pragma unroll
    for (int m = 1; m < 32; m <<= 1) s2 += __shfl_xor(s2, m, 32);
    float rn = 1.0f / fmaxf(sqrtf(s2), 1e-12f);
    if (lane < DR) out[(long)row * DR + lane] = kv * rn;
}

// ---------------------------------------------------------------------------
// ATD cross-attention vs token dictionary: one wave per token.
// ---------------------------------------------------------------------------
__global__ __launch_bounds__(256) void atd_kernel(
    const __bf16* __restrict__ xn, const float* __restrict__ wq, const float* __restrict__ qb,
    const float* __restrict__ knorm, const float* __restrict__ scale1,
    __bf16* __restrict__ sim, int* __restrict__ tk)
{
    const int b = blockIdx.y;
    const int wid = threadIdx.x >> 5, lane = threadIdx.x & 31;
    const int t = blockIdx.x * 8 + wid;
    __shared__ float kn_s[Mtd * DR];
    __shared__ float xs[8][DIM];
    __shared__ float qs[8][DR];
    for (int i = threadIdx.x; i < Mtd * DR; i += 256) kn_s[i] = knorm[(long)b * Mtd * DR + i];
    const __bf16* xp = xn + ((long)b * Nn + t) * DIMP;
    for (int i = lane; i < DIM; i += 32) xs[wid][i] = (float)xp[i];
    __syncthreads();
    float qv = 0.f;
    if (lane < DR) {
        for (int c = 0; c < DIM; ++c) qv += xs[wid][c] * wq[c * DR + lane];
        qv += qb[lane];
    }
    float q2 = (lane < DR) ? qv * qv : 0.f;
    #pragma unroll
    for (int m = 1; m < 32; m <<= 1) q2 += __shfl_xor(q2, m, 32);
    float rn = 1.0f / fmaxf(sqrtf(q2), 1e-12f);
    if (lane < DR) qs[wid][lane] = qv * rn;
    __syncthreads();
    float sc = fminf(fmaxf(scale1[0], 0.f), 3.f);
    float mult = 1.0f + sc * logf((float)Mtd);
    float lo[4];
    #pragma unroll
    for (int mi = 0; mi < 4; ++mi) {
        int m = mi * 32 + lane;
        float d = 0.f;
        #pragma unroll
        for (int j = 0; j < DR; ++j) d += qs[wid][j] * kn_s[m * DR + j];
        lo[mi] = d * mult;
    }
    float mx = lo[0]; int arg = lane;
    #pragma unroll
    for (int mi = 1; mi < 4; ++mi) if (lo[mi] > mx) { mx = lo[mi]; arg = mi * 32 + lane; }
    float bm = mx; int ba = arg;
    #pragma unroll
    for (int m = 1; m < 32; m <<= 1) {
        float om = __shfl_xor(bm, m, 32);
        int   oa = __shfl_xor(ba, m, 32);
        if (om > bm || (om == bm && oa < ba)) { bm = om; ba = oa; }
    }
    float gmx = mx;
    #pragma unroll
    for (int m = 1; m < 32; m <<= 1) gmx = fmaxf(gmx, __shfl_xor(gmx, m, 32));
    float se = 0.f;
    #pragma unroll
    for (int mi = 0; mi < 4; ++mi) { lo[mi] = __expf(lo[mi] - gmx); se += lo[mi]; }
    #pragma unroll
    for (int m = 1; m < 32; m <<= 1) se += __shfl_xor(se, m, 32);
    float rinv = 1.0f / se;
    __bf16* sp = sim + ((long)b * Nn + t) * Mtd;
    #pragma unroll
    for (int mi = 0; mi < 4; ++mi) sp[mi * 32 + lane] = (__bf16)(lo[mi] * rinv);
    if (lane == 0) tk[(long)b * Nn + t] = ba;
}

// ---------------------------------------------------------------------------
// Counting sort of tokens by category.
// ---------------------------------------------------------------------------
__global__ void hist_kernel(const int* __restrict__ tk, int* __restrict__ counts)
{
    int i = blockIdx.x * 256 + threadIdx.x;
    int b = i >> 14;
    atomicAdd(&counts[b * Mtd + tk[i]], 1);
}
__global__ void scan_kernel(const int* __restrict__ counts, int* __restrict__ cursor)
{
    int b = threadIdx.x;
    if (b >= Bn) return;
    int acc = 0;
    for (int c = 0; c < Mtd; ++c) { cursor[b * Mtd + c] = acc; acc += counts[b * Mtd + c]; }
}
__global__ void scatter_kernel(const int* __restrict__ tk, int* __restrict__ cursor,
                               int* __restrict__ sidx)
{
    int i = blockIdx.x * 256 + threadIdx.x;
    int b = i >> 14, t = i & (Nn - 1);
    int pos = atomicAdd(&cursor[b * Mtd + tk[i]], 1);
    sidx[(long)b * Nn + pos] = t;
}

// ---------------------------------------------------------------------------
// 256-token attention; S kept in registers, K^T and V staged swizzled in LDS.
// MODE 0: category groups (sorted index); MODE 1: shifted windows + bias+mask.
// ---------------------------------------------------------------------------
template<int MODE>
__global__ __launch_bounds__(256) void attn256(
    const __bf16* __restrict__ qkv, const int* __restrict__ sidx,
    const float* __restrict__ rpb, const int* __restrict__ rpi,
    const float* __restrict__ amask, __bf16* __restrict__ outh)
{
    __shared__ char smem[57344];
    __shared__ float pmax[8][64];
    __shared__ float psum[8][64];
    __shared__ float rmax[64];
    __shared__ float rsum[64];
    const int tid = threadIdx.x, lane = tid & 31, wid = tid >> 5;
    const int l4 = lane & 15, lh = lane >> 4;
    int cx = blockIdx.x;
    const int head = cx % HEADS; cx /= HEADS;
    const int g = cx & 63;
    const int b = cx >> 6;
    const int qt = blockIdx.y;
    const long qbase = (long)b * Nn * C3;

    auto map_tok = [&](int j) -> int {
        if (MODE == 0) return sidx[(long)b * Nn + g * 256 + j];
        int wh = g >> 3, wc = g & 7;
        int r = j >> 4, c = j & 15;
        return (((wh * 16 + r + 8) & 127) * Wwid) + ((wc * 16 + c + 8) & 127);
    };

    __bf16* Qs  = (__bf16*)smem;              // [64][64] natural
    __bf16* Ksw = (__bf16*)(smem + 8192);     // swizzled: [(kk*16+nt)*32+lane][16]
    const float qscale = rsqrtf((float)HD);
    for (int i = tid; i < 64 * 64; i += 256) {
        int m = i >> 6, k = i & 63;
        float v = 0.0f;
        if (k < HD) {
            int t = map_tok(qt * 64 + m);
            v = (float)qkv[qbase + (long)t * C3 + head * HD + k] * qscale;
        }
        Qs[i] = (__bf16)v;
    }
    for (int i = tid; i < 256 * 64; i += 256) {
        int n = i >> 6, k = i & 63;
        float v = 0.0f;
        if (k < HD) {
            int t = map_tok(n);
            v = (float)qkv[qbase + (long)t * C3 + DIM + head * HD + k];
        }
        Ksw[bswz_addr((k >> 5) * 16 + (n >> 4), k & 31, n & 15)] = (__bf16)v;
    }
    __syncthreads();

    v8f S[4][2] = {};
    #pragma unroll
    for (int mt = 0; mt < 4; ++mt)
        #pragma unroll
        for (int nt = 0; nt < 2; ++nt)
            #pragma unroll
            for (int kk = 0; kk < 2; ++kk) {
                v16bf af = frag_a(&Qs[(mt * 16 + l4) * 64 + kk * 32], lh);
                v16bf bfr = frag_b_sw(Ksw, kk * 16 + (wid * 2 + nt), lane);
                S[mt][nt] = __builtin_amdgcn_wmma_f32_16x16x32_bf16(false, af, false, bfr, (short)0, S[mt][nt], false, false);
            }
    if (MODE == 1) {
        #pragma unroll
        for (int mt = 0; mt < 4; ++mt)
            #pragma unroll
            for (int nt = 0; nt < 2; ++nt)
                #pragma unroll
                for (int r = 0; r < 8; ++r) {
                    int qs_ = qt * 64 + mt * 16 + lh * 8 + r;
                    int ks_ = wid * 32 + nt * 16 + l4;
                    S[mt][nt][r] += rpb[rpi[qs_ * 256 + ks_] * HEADS + head]
                                  + amask[((long)g * 256 + qs_) * 256 + ks_];
                }
    }
    // ---- cross-wave softmax over 256 columns ----
    #pragma unroll
    for (int mt = 0; mt < 4; ++mt)
        #pragma unroll
        for (int r = 0; r < 8; ++r) {
            float v = fmaxf(S[mt][0][r], S[mt][1][r]);
            #pragma unroll
            for (int m = 1; m < 16; m <<= 1) v = fmaxf(v, __shfl_xor(v, m, 32));
            if (l4 == 0) pmax[wid][mt * 16 + lh * 8 + r] = v;
        }
    __syncthreads();
    if (tid < 64) {
        float v = pmax[0][tid];
        #pragma unroll
        for (int w2 = 1; w2 < 8; ++w2) v = fmaxf(v, pmax[w2][tid]);
        rmax[tid] = v;
    }
    __syncthreads();
    #pragma unroll
    for (int mt = 0; mt < 4; ++mt)
        #pragma unroll
        for (int r = 0; r < 8; ++r) {
            int rowi = mt * 16 + lh * 8 + r;
            float rm = rmax[rowi];
            S[mt][0][r] = __expf(S[mt][0][r] - rm);
            S[mt][1][r] = __expf(S[mt][1][r] - rm);
            float s = S[mt][0][r] + S[mt][1][r];
            #pragma unroll
            for (int m = 1; m < 16; m <<= 1) s += __shfl_xor(s, m, 32);
            if (l4 == 0) psum[wid][rowi] = s;
        }
    __syncthreads();
    if (tid < 64) {
        float s = 0.f;
        #pragma unroll
        for (int w2 = 0; w2 < 8; ++w2) s += psum[w2][tid];
        rsum[tid] = 1.0f / s;
    }
    __syncthreads();
    // ---- write P (bf16, over dead Q/K LDS) and stage V swizzled ----
    __bf16* Ps  = (__bf16*)smem;              // [64][256] natural (A operand)
    __bf16* Vsw = (__bf16*)(smem + 32768);    // swizzled: [(kb*3+nt)*32+lane][16]
    #pragma unroll
    for (int mt = 0; mt < 4; ++mt)
        #pragma unroll
        for (int nt = 0; nt < 2; ++nt)
            #pragma unroll
            for (int r = 0; r < 8; ++r) {
                int rowi = mt * 16 + lh * 8 + r;
                int coli = wid * 32 + nt * 16 + l4;
                Ps[rowi * 256 + coli] = (__bf16)(S[mt][nt][r] * rsum[rowi]);
            }
    for (int i = tid; i < 256 * 48; i += 256) {
        int n = i / 48, c = i % 48;            // n = K index, c = head-dim col
        float v = 0.0f;
        if (c < HD) {
            int t = map_tok(n);
            v = (float)qkv[qbase + (long)t * C3 + 2 * DIM + head * HD + c];
        }
        Vsw[bswz_addr((n >> 5) * 3 + (c >> 4), n & 31, c & 15)] = (__bf16)v;
    }
    __syncthreads();
    // ---- O = P @ V  (64x48, K=256) ----
    for (int tt = wid; tt < 12; tt += 8) {
        int mt = tt & 3, ntile = tt >> 2;
        v8f o = {};
        #pragma unroll
        for (int kb = 0; kb < 8; ++kb) {
            v16bf af = frag_a(&Ps[(mt * 16 + l4) * 256 + kb * 32], lh);
            v16bf bfr = frag_b_sw(Vsw, kb * 3 + ntile, lane);
            o = __builtin_amdgcn_wmma_f32_16x16x32_bf16(false, af, false, bfr, (short)0, o, false, false);
        }
        int col = ntile * 16 + l4;
        if (col < HD) {
            #pragma unroll
            for (int r = 0; r < 8; ++r) {
                int t = map_tok(qt * 64 + mt * 16 + lh * 8 + r);
                outh[((long)b * Nn + t) * DIMP + head * HD + col] = (__bf16)o[r];
            }
        }
    }
}

// ---------------------------------------------------------------------------
// Gather per-token dictionary features into planar channels 420..515.
// ---------------------------------------------------------------------------
__global__ void gather_td_kernel(const float* __restrict__ ftd, const int* __restrict__ tk,
                                 float* __restrict__ planar)
{
    int n = blockIdx.x * 256 + threadIdx.x;
    int j = blockIdx.y, b = blockIdx.z;
    int cat = tk[(long)b * Nn + n];
    planar[((long)b * HF2 + (MLP + j)) * Nn + n] = ftd[((long)b * Mtd + cat) * DTD + j];
}

// ---------------------------------------------------------------------------
// Depthwise 5x5 conv + bias + GELU + residual; repack token-major bf16 [.,544].
// ---------------------------------------------------------------------------
__global__ __launch_bounds__(256) void dwconv_kernel(const float* __restrict__ planar,
                                                     const float* __restrict__ w5,
                                                     const float* __restrict__ cb,
                                                     __bf16* __restrict__ out)
{
    const int tile = blockIdx.x, ch = blockIdx.y, b = blockIdx.z;
    const int tr = (tile >> 3) * 16, tc = (tile & 7) * 16;
    __shared__ float tS[20][20];
    __shared__ float wS[25];
    const float* ip = planar + ((long)b * HF2 + ch) * Nn;
    for (int i = threadIdx.x; i < 400; i += 256) {
        int r = i / 20, c = i % 20;
        int gr = tr + r - 2, gc = tc + c - 2;
        tS[r][c] = (gr >= 0 && gr < Hh && gc >= 0 && gc < Wwid) ? ip[gr * Wwid + gc] : 0.f;
    }
    if (threadIdx.x < 25) wS[threadIdx.x] = w5[ch * 25 + threadIdx.x];
    __syncthreads();
    int r = threadIdx.x >> 4, c = threadIdx.x & 15;
    float s = 0.f;
    #pragma unroll
    for (int i = 0; i < 5; ++i)
        #pragma unroll
        for (int j = 0; j < 5; ++j)
            s += tS[r + i][c + j] * wS[i * 5 + j];
    s += cb[ch];
    float v = tS[r + 2][c + 2] + geluf(s);
    int n = (tr + r) * Wwid + tc + c;
    out[((long)b * Nn + n) * HF2P + ch] = (__bf16)v;
}

// ===========================================================================
extern "C" void kernel_launch(void* const* d_in, const int* in_sizes, int n_in,
                              void* d_out, int out_size, void* d_ws, size_t ws_size,
                              hipStream_t stream)
{
    (void)in_sizes; (void)n_in; (void)out_size; (void)ws_size;
    const float* x      = (const float*)d_in[0];
    const float* td     = (const float*)d_in[1];
    const float* g1     = (const float*)d_in[2];
    const float* b1     = (const float*)d_in[3];
    const float* g2     = (const float*)d_in[4];
    const float* b2     = (const float*)d_in[5];
    const float* wqkv_w = (const float*)d_in[6];
    const float* wqkv_b = (const float*)d_in[7];
    const float* wq_w   = (const float*)d_in[8];
    const float* wq_b   = (const float*)d_in[9];
    const float* wk_w   = (const float*)d_in[10];
    const float* wk_b   = (const float*)d_in[11];
    const float* wv_w   = (const float*)d_in[12];
    const float* wv_b   = (const float*)d_in[13];
    const float* atds   = (const float*)d_in[14];
    const float* aca_w  = (const float*)d_in[15];
    const float* aca_b  = (const float*)d_in[16];
    const float* rpb    = (const float*)d_in[17];
    const float* win_w  = (const float*)d_in[18];
    const float* win_b  = (const float*)d_in[19];
    const float* fctd_w = (const float*)d_in[20];
    const float* fctd_b = (const float*)d_in[21];
    const float* fc1_w  = (const float*)d_in[22];
    const float* fc1_b  = (const float*)d_in[23];
    const float* dw_w   = (const float*)d_in[24];
    const float* dw_b   = (const float*)d_in[25];
    const float* fc2_w  = (const float*)d_in[26];
    const float* fc2_b  = (const float*)d_in[27];
    const int*   rpi    = (const int*)d_in[28];
    const float* amask  = (const float*)d_in[29];
    float* out = (float*)d_out;

    char* wsp = (char*)d_ws;
    size_t off = 0;
    auto alloc = [&](size_t bytes) -> void* {
        void* p = wsp + off;
        off = (off + bytes + 255) & ~((size_t)255);
        return p;
    };
    __bf16* wqkv_h = (__bf16*)alloc((size_t)DIMP * C3N * 2);
    __bf16* wv_h   = (__bf16*)alloc((size_t)DIMP * DIMN * 2);
    __bf16* aca_h  = (__bf16*)alloc((size_t)DIMP * DIMN * 2);
    __bf16* win_h  = (__bf16*)alloc((size_t)DIMP * DIMN * 2);
    __bf16* fctd_h = (__bf16*)alloc((size_t)DIMP * DTDN * 2);
    __bf16* fc1_h  = (__bf16*)alloc((size_t)DIMP * MLPN * 2);
    __bf16* fc2_h  = (__bf16*)alloc((size_t)HF2P * DIMN * 2);
    __bf16* td_h   = (__bf16*)alloc((size_t)Bn * Mtd * DIMP * 2);
    __bf16* xn_h   = (__bf16*)alloc((size_t)Bn * Nn * DIMP * 2);
    __bf16* xn2_h  = (__bf16*)alloc((size_t)Bn * Nn * DIMP * 2);
    __bf16* qkv_h  = (__bf16*)alloc((size_t)Bn * Nn * C3 * 2);
    __bf16* sim_h  = (__bf16*)alloc((size_t)Bn * Nn * Mtd * 2);
    int*    tk     = (int*)   alloc((size_t)Bn * Nn * 4);
    float*  knormb = (float*) alloc((size_t)Bn * Mtd * DR * 4);
    __bf16* vbuf   = (__bf16*)alloc((size_t)Bn * Mtd * DIMN * 2);
    int*    counts = (int*)   alloc((size_t)Bn * Mtd * 4);
    int*    cursor = (int*)   alloc((size_t)Bn * Mtd * 4);
    int*    sidx   = (int*)   alloc((size_t)Bn * Nn * 4);
    float*  acc    = (float*) alloc((size_t)Bn * Nn * DIM * 4);
    __bf16* acah   = (__bf16*)alloc((size_t)Bn * Nn * DIMP * 2);
    __bf16* winh   = (__bf16*)alloc((size_t)Bn * Nn * DIMP * 2);
    float*  ftd    = (float*) alloc((size_t)Bn * Mtd * DTD * 4);
    float*  planar = (float*) alloc((size_t)Bn * HF2 * Nn * 4);
    __bf16* hcat2  = (__bf16*)alloc((size_t)Bn * Nn * HF2P * 2);

    auto cvt = [&](const float* s, __bf16* d, int rows, int cols, int ldd, int rowsPad) {
        long tot = (long)rowsPad * ldd;
        cvt_pad_kernel<<<(int)((tot + 255) / 256), 256, 0, stream>>>(s, d, rows, cols, ldd, rowsPad);
    };
    cvt(wqkv_w, wqkv_h, DIM, C3, C3N, DIMP);
    cvt(wv_w,   wv_h,   DIM, DIM, DIMN, DIMP);
    cvt(aca_w,  aca_h,  DIM, DIM, DIMN, DIMP);
    cvt(win_w,  win_h,  DIM, DIM, DIMN, DIMP);
    cvt(fctd_w, fctd_h, DIM, DTD, DTDN, DIMP);
    cvt(fc1_w,  fc1_h,  DIM, MLP, MLPN, DIMP);
    cvt(fc2_w,  fc2_h,  HF2, DIM, DIMN, HF2P);
    cvt(td,     td_h,   Bn * Mtd, DIM, DIMP, Bn * Mtd);

    ln_kernel<<<Bn * Nn / 8, 256, 0, stream>>>(x, g1, b1, xn_h, Bn * Nn);
    knorm_kernel<<<Bn * Mtd / 8, 256, 0, stream>>>(td, wk_w, wk_b, knormb);

    // qkv = xn @ wqkv + b  (bf16 out [B,N,630])
    gemm_bf16_wmma<true, false, false, false><<<dim3(10, Nn / 64, Bn), 256, 0, stream>>>(
        xn_h, (long)Nn * DIMP, DIMP, wqkv_h, 0, C3N, wqkv_b,
        nullptr, 0, 0, nullptr, qkv_h, (long)Nn * C3, C3, Nn, C3, DIMP);

    atd_kernel<<<dim3(Nn / 8, Bn), 256, 0, stream>>>(xn_h, wq_w, wq_b, knormb, atds, sim_h, tk);

    // v = td @ wv + b  (bf16, used as batched W [128 x 256-padded])
    gemm_bf16_wmma<true, false, false, false><<<dim3(4, Mtd / 64, Bn), 256, 0, stream>>>(
        td_h, (long)Mtd * DIMP, DIMP, wv_h, 0, DIMN, wv_b,
        nullptr, 0, 0, nullptr, vbuf, (long)Mtd * DIMN, DIMN, Mtd, DIM, DIMP);

    hipMemcpyAsync(acc, x, sizeof(float) * (size_t)Bn * Nn * DIM, hipMemcpyDeviceToDevice, stream);

    // acc += sim @ v
    gemm_bf16_wmma<false, true, false, false><<<dim3(4, Nn / 64, Bn), 256, 0, stream>>>(
        sim_h, (long)Nn * Mtd, Mtd, vbuf, (long)Mtd * DIMN, DIMN, nullptr,
        acc, (long)Nn * DIM, DIM, acc, nullptr, (long)Nn * DIM, DIM, Nn, DIM, Mtd);

    hipMemsetAsync(counts, 0, (size_t)Bn * Mtd * 4, stream);
    hist_kernel<<<Bn * Nn / 256, 256, 0, stream>>>(tk, counts);
    scan_kernel<<<1, 32, 0, stream>>>(counts, cursor);
    scatter_kernel<<<Bn * Nn / 256, 256, 0, stream>>>(tk, cursor, sidx);

    attn256<0><<<dim3(Bn * 64 * HEADS, 4), 256, 0, stream>>>(qkv_h, sidx, rpb, rpi, amask, acah);
    attn256<1><<<dim3(Bn * 64 * HEADS, 4), 256, 0, stream>>>(qkv_h, sidx, rpb, rpi, amask, winh);

    gemm_bf16_wmma<false, true, false, false><<<dim3(4, Nn / 64, Bn), 256, 0, stream>>>(
        acah, (long)Nn * DIMP, DIMP, aca_h, 0, DIMN, aca_b,
        acc, (long)Nn * DIM, DIM, acc, nullptr, (long)Nn * DIM, DIM, Nn, DIM, DIMP);
    gemm_bf16_wmma<false, true, false, false><<<dim3(4, Nn / 64, Bn), 256, 0, stream>>>(
        winh, (long)Nn * DIMP, DIMP, win_h, 0, DIMN, win_b,
        acc, (long)Nn * DIM, DIM, acc, nullptr, (long)Nn * DIM, DIM, Nn, DIM, DIMP);

    // ftd = td @ fc_td + b
    gemm_bf16_wmma<false, false, false, false><<<dim3(2, Mtd / 64, Bn), 256, 0, stream>>>(
        td_h, (long)Mtd * DIMP, DIMP, fctd_h, 0, DTDN, fctd_b,
        nullptr, 0, 0, ftd, nullptr, (long)Mtd * DTD, DTD, Mtd, DTD, DIMP);

    ln_kernel<<<Bn * Nn / 8, 256, 0, stream>>>(acc, g2, b2, xn2_h, Bn * Nn);

    // planar[ch<420] = gelu(xn2 @ fc1 + b)  (transposed / NCHW-planar store)
    gemm_bf16_wmma<false, false, true, true><<<dim3(7, Nn / 64, Bn), 256, 0, stream>>>(
        xn2_h, (long)Nn * DIMP, DIMP, fc1_h, 0, MLPN, fc1_b,
        nullptr, 0, 0, planar, nullptr, (long)HF2 * Nn, Nn, Nn, MLP, DIMP);

    gather_td_kernel<<<dim3(Nn / 256, DTD, Bn), 256, 0, stream>>>(ftd, tk, planar);

    dwconv_kernel<<<dim3(64, HF2, Bn), 256, 0, stream>>>(planar, dw_w, dw_b, hcat2);

    // out = acc + hcat @ fc2 + b
    gemm_bf16_wmma<false, true, false, false><<<dim3(4, Nn / 64, Bn), 256, 0, stream>>>(
        hcat2, (long)Nn * HF2P, HF2P, fc2_h, 0, DIMN, fc2_b,
        acc, (long)Nn * DIM, DIM, out, nullptr, (long)Nn * DIM, DIM, Nn, DIM, HF2P);
}
